// LSTMModel_51204600103141
// MI455X (gfx1250) — matrix operand order repaired and checked by reading, not compiled, not run on hardware
//
#include <hip/hip_runtime.h>

#define NODES 19
#define DIN   100
#define H     128
#define TSTEP 250
#define BATCH 64
#define NCLS  4
#define GDIM  512          // 4*H gate rows
#define KTOT  256          // padded K: [x_t pad128 | h 128]
#define WPITCH 264         // halves per LDS row (pad to break bank conflicts)
#define NTHR1 256          // 8 waves -> 4 waves/EU -> double per-wave VGPR budget
#define FEAT  (NODES*H)    // 2432

typedef __attribute__((ext_vector_type(16))) __bf16        v16bf;
typedef __attribute__((ext_vector_type(8)))  float         v8f;
typedef __attribute__((ext_vector_type(4)))  float         v4f;
typedef __attribute__((ext_vector_type(4)))  unsigned int  v4u;
typedef __attribute__((ext_vector_type(2)))  unsigned int  v2u;

union Frag {
  v16bf bf;
  v4u   q[2];
  unsigned short u[16];
};

__device__ __forceinline__ unsigned short f2bf(float f) {
  unsigned int u = __float_as_uint(f);
  u += 0x7FFFu + ((u >> 16) & 1u);            // round-to-nearest-even
  return (unsigned short)(u >> 16);
}
__device__ __forceinline__ float sigm(float x)  { return 1.0f / (1.0f + __expf(-x)); }
__device__ __forceinline__ float tanh_f(float x){ float e = __expf(-2.0f * x); return (1.0f - e) / (1.0f + e); }
__device__ __forceinline__ v8f zero8() { v8f z = {0.f,0.f,0.f,0.f,0.f,0.f,0.f,0.f}; return z; }

// One persistent workgroup per node. LDS: W_cat bf16 [512][WPITCH], A_cat bf16 [64][WPITCH], bias f32[512].
// 8 waves; wave w owns batch-tile (w>>1) and j-tiles (w&1)*4 .. (w&1)*4+3  (16 gate tiles, acc=128 VGPRs).
extern "C" __global__ __launch_bounds__(NTHR1)
__attribute__((amdgpu_waves_per_eu(4)))
void lstm_persistent(const float* __restrict__ x,      // [B][NODES][T][DIN]
                     const float* __restrict__ W_ih,   // [NODES][512][100]
                     const float* __restrict__ W_hh,   // [NODES][512][128]
                     const float* __restrict__ b_ih,   // [NODES][512]
                     const float* __restrict__ b_hh,   // [NODES][512]
                     unsigned short* __restrict__ hbf) // out: [B][FEAT] bf16
{
  extern __shared__ unsigned char smem[];
  unsigned short* Wl    = (unsigned short*)smem;        // [GDIM][WPITCH]
  unsigned short* Al    = Wl + GDIM * WPITCH;           // [BATCH][WPITCH]
  float*          biasl = (float*)(Al + BATCH * WPITCH);// [GDIM]

  const int n    = blockIdx.x;
  const int tid  = threadIdx.x;
  const int lane = tid & 31;
  const int wave = tid >> 5;              // 0..7
  const int hi   = (lane & 16) ? 1 : 0;
  const int l15  = lane & 15;
  const int btile = wave >> 1;            // 0..3
  const int jt0   = (wave & 1) * 4;       // first of 4 consecutive j-tiles

  // ---- one-time staging: W_cat = [W_ih (pad K->128) | W_hh] as bf16 ----
  for (int idx = tid; idx < GDIM * 128; idx += NTHR1) {
    int g = idx >> 7, k = idx & 127;
    float w = (k < DIN) ? W_ih[(n * GDIM + g) * DIN + k] : 0.0f;
    Wl[g * WPITCH + k] = f2bf(w);
  }
  for (int idx = tid; idx < GDIM * 128; idx += NTHR1) {
    int g = idx >> 7, k = idx & 127;
    Wl[g * WPITCH + 128 + k] = f2bf(W_hh[(n * GDIM + g) * H + k]);
  }
  for (int idx = tid; idx < GDIM; idx += NTHR1)
    biasl[idx] = b_ih[n * GDIM + idx] + b_hh[n * GDIM + idx];
  // zero x-pad cols (100..127) and h cols (128..255 -> h0 = 0)
  for (int idx = tid; idx < BATCH * KTOT; idx += NTHR1) {
    int b = idx >> 8, k = idx & 255;
    if (k >= DIN) Al[b * WPITCH + k] = 0;
  }

  // ---- fragment base pointers (loop-invariant) ----
  // ISA 16-bit A layout: lanes<16 hold K{0..7,16..23}, lanes>=16 K{8..15,24..31}
  const unsigned short* aBase = &Al[(btile * 16 + l15) * WPITCH + hi * 8];
  // B layout: lanes<16 K{0..7,16..23}, lanes>=16 K{8..15,24..31}. One base per gate; the 4 j-tiles
  // are consecutive -> immediate offsets pi*16*WPITCH*2 = pi*8448 B (fits DS imm).
  const unsigned short* bGate[4];
  #pragma unroll
  for (int g = 0; g < 4; ++g)
    bGate[g] = &Wl[(g * H + jt0 * 16 + l15) * WPITCH + hi * 8];

  v8f creg[4];
  #pragma unroll
  for (int pi = 0; pi < 4; ++pi) creg[pi] = zero8();   // cell state lives in VGPRs

  #pragma unroll 1
  for (int t = 0; t < TSTEP; ++t) {
    // stage x_t (fp32 -> bf16), vectorized: 25 float4 per batch row
    for (int idx = tid; idx < BATCH * 25; idx += NTHR1) {
      const int b = idx / 25, q = idx - b * 25;
      const v4f xv = *(const v4f*)&x[((size_t)(b * NODES + n) * TSTEP + t) * DIN + q * 4];
      v2u pk;
      pk[0] = (unsigned int)f2bf(xv[0]) | ((unsigned int)f2bf(xv[1]) << 16);
      pk[1] = (unsigned int)f2bf(xv[2]) | ((unsigned int)f2bf(xv[3]) << 16);
      *(v2u*)&Al[b * WPITCH + q * 4] = pk;
    }
    // prefetch next timestep's rows (4 touches per 400 B row, 100 B apart)
    if (t + 1 < TSTEP && tid < BATCH * 4) {
      const int b = tid >> 2, c = tid & 3;
      __builtin_prefetch(&x[((size_t)(b * NODES + n) * TSTEP + (t + 1)) * DIN + c * 25], 0, 0);
    }
    __syncthreads();

    // gates[64,512] = A_cat[64,256] @ W_catT[256,512]: 8 K-chunks x 16 tiles/wave.
    // 2-deep pipeline: b_nxt loads 2 DS-ops ahead of each WMMA; a_nxt a chunk ahead.
    v8f acc[16];
    #pragma unroll
    for (int u = 0; u < 16; ++u) acc[u] = zero8();

    Frag a_cur, a_nxt, b_cur, b_nxt;
    a_cur.q[0] = *(const v4u*)(aBase);
    a_cur.q[1] = *(const v4u*)(aBase + 16);
    b_cur.q[0] = *(const v4u*)(bGate[0]);
    b_cur.q[1] = *(const v4u*)(bGate[0] + 16);

    #pragma unroll
    for (int kc = 0; kc < 8; ++kc) {
      const int k0 = kc * 32;           // halves
      const int k1 = (kc < 7) ? k0 + 32 : k0;
      if (kc < 7) {                     // chunk kc+1's A, loaded a whole chunk early
        a_nxt.q[0] = *(const v4u*)(aBase + k1);
        a_nxt.q[1] = *(const v4u*)(aBase + k1 + 16);
      }
      #pragma unroll
      for (int u = 0; u < 16; ++u) {    // u = pi*4 + g
        if (u < 15) {
          const int un = u + 1;
          const unsigned short* bp = bGate[un & 3] + (un >> 2) * (16 * WPITCH) + k0;
          b_nxt.q[0] = *(const v4u*)bp;
          b_nxt.q[1] = *(const v4u*)(bp + 16);
        } else {
          const unsigned short* bp = bGate[0] + k1;
          b_nxt.q[0] = *(const v4u*)bp;
          b_nxt.q[1] = *(const v4u*)(bp + 16);
        }
        acc[u] = __builtin_amdgcn_wmma_f32_16x16x32_bf16(
            false, a_cur.bf, false, b_cur.bf, (short)0, acc[u], false, false);
        b_cur = b_nxt;                  // unrolled -> pure register renaming
      }
      if (kc < 7) a_cur = a_nxt;
    }
    __syncthreads();   // all matmul reads done before h is overwritten

    // fused LSTM cell update (fp32); h -> A_cat cols 128.. as bf16; c stays in VGPRs
    #pragma unroll
    for (int pi = 0; pi < 4; ++pi) {
      const int ncol = (jt0 + pi) * 16 + l15;
      const float bi = biasl[ncol],         bf_ = biasl[H + ncol];
      const float bg = biasl[2 * H + ncol], bo  = biasl[3 * H + ncol];
      #pragma unroll
      for (int r = 0; r < 8; ++r) {
        float gi = sigm  (acc[pi * 4 + 0][r] + bi);
        float gf = sigm  (acc[pi * 4 + 1][r] + bf_);
        float gg = tanh_f(acc[pi * 4 + 2][r] + bg);
        float go = sigm  (acc[pi * 4 + 3][r] + bo);
        float cv = gf * creg[pi][r] + gi * gg;
        creg[pi][r] = cv;
        float hv = go * tanh_f(cv);
        const int b = btile * 16 + r + hi * 8;   // batch row (C-tile M index)
        Al[b * WPITCH + 128 + ncol] = f2bf(hv);
      }
    }
    // next-iter x staging only touches cols <128; barrier at loop top guards matmul
  }

  __syncthreads();   // final h visible to all waves
  for (int idx = tid; idx < BATCH * H; idx += NTHR1) {
    const int b = idx >> 7, j = idx & 127;
    hbf[b * FEAT + n * H + j] = Al[b * WPITCH + 128 + j];
  }
}

// out = relu(hcat @ W1^T + b1) @ W2^T + b2 ; hcat read as bf16, W1 converted in-register
extern "C" __global__ __launch_bounds__(256)
void classifier(const unsigned short* __restrict__ hbf,  // [B][FEAT] bf16
                const float* __restrict__ W1, const float* __restrict__ b1,
                const float* __restrict__ W2, const float* __restrict__ b2,
                float* __restrict__ out)
{
  __shared__ float hid[BATCH][H + 4];
  const int tid  = threadIdx.x;
  const int lane = tid & 31;
  const int wave = tid >> 5;
  const int hi   = (lane & 16) ? 1 : 0;
  const int l15  = lane & 15;
  const int btile = wave & 3;
  const int jbase = (wave >> 2) * 4;     // 4 jtiles per wave

  v8f acc[4];
  #pragma unroll
  for (int jt = 0; jt < 4; ++jt) acc[jt] = zero8();

  const unsigned short* aBase = &hbf[(btile * 16 + l15) * FEAT + hi * 8];
  const float* wBase[4];
  #pragma unroll
  for (int jt = 0; jt < 4; ++jt)
    wBase[jt] = &W1[(size_t)((jbase + jt) * 16 + l15) * FEAT + hi * 8];

  for (int kc = 0; kc < FEAT / 32; ++kc) {   // 76 chunks, K = 2432 exactly
    const int k0 = kc * 32;
    Frag a;
    a.q[0] = *(const v4u*)(aBase + k0);
    a.q[1] = *(const v4u*)(aBase + k0 + 16);
    #pragma unroll
    for (int jt = 0; jt < 4; ++jt) {
      const float* wp = wBase[jt] + k0;
      Frag bb;
      #pragma unroll
      for (int i = 0; i < 16; ++i) bb.u[i] = f2bf(wp[(i & 7) + ((i >> 3) << 4)]);
      acc[jt] = __builtin_amdgcn_wmma_f32_16x16x32_bf16(
          false, a.bf, false, bb.bf, (short)0, acc[jt], false, false);
    }
  }

  #pragma unroll
  for (int jt = 0; jt < 4; ++jt) {
    const int ncol = (jbase + jt) * 16 + l15;
    const float bia = b1[ncol];
    #pragma unroll
    for (int r = 0; r < 8; ++r) {
      float v = acc[jt][r] + bia;
      hid[btile * 16 + r + hi * 8][ncol] = v > 0.f ? v : 0.f;   // ReLU
    }
  }
  __syncthreads();

  // tiny layer 2: 256 threads = (b, cls) pairs
  const int b = tid >> 2, cls = tid & 3;
  float s = b2[cls];
  #pragma unroll 4
  for (int k = 0; k < H; ++k) s += hid[b][k] * W2[cls * H + k];
  out[b * NCLS + cls] = s;
}

extern "C" void kernel_launch(void* const* d_in, const int* in_sizes, int n_in,
                              void* d_out, int out_size, void* d_ws, size_t ws_size,
                              hipStream_t stream) {
  (void)in_sizes; (void)n_in; (void)out_size; (void)ws_size;
  const float* x    = (const float*)d_in[0];
  // d_in[1] = seq_lengths (int64): unused — the reference scan ignores it.
  const float* W_ih = (const float*)d_in[2];
  const float* W_hh = (const float*)d_in[3];
  const float* b_ih = (const float*)d_in[4];
  const float* b_hh = (const float*)d_in[5];
  const float* W1   = (const float*)d_in[6];
  const float* b1   = (const float*)d_in[7];
  const float* W2   = (const float*)d_in[8];
  const float* b2   = (const float*)d_in[9];
  float* out = (float*)d_out;
  unsigned short* hbf = (unsigned short*)d_ws;   // 64*2432 bf16 = 311,296 B

  const size_t smem = (size_t)(GDIM + BATCH) * WPITCH * sizeof(unsigned short)
                    + (size_t)GDIM * sizeof(float);   // 306,176 B < 320 KB
  hipFuncSetAttribute(reinterpret_cast<const void*>(lstm_persistent),
                      hipFuncAttributeMaxDynamicSharedMemorySize, (int)smem);

  lstm_persistent<<<dim3(NODES), dim3(NTHR1), smem, stream>>>(
      x, W_ih, W_hh, b_ih, b_hh, hbf);
  classifier<<<dim3(1), dim3(256), 0, stream>>>(hbf, W1, b1, W2, b2, out);
}
